// MultiheadAttention_22789096473211
// MI455X (gfx1250) — matrix operand order repaired and checked by reading, not compiled, not run on hardware
//
#include <hip/hip_runtime.h>

// MHA forward for MI455X (gfx1250, wave32, WMMA 16x16x32 bf16, f32 accum).
// Pipeline: [QKV gemms: f32 in -> bf16 out] -> [flash attention] -> [out proj f32].
// CDNA5 paths: v_wmma_f32_16x16x32_bf16, tensor_load_to_lds (TENSORcnt, TDM),
// global_load_async_to_lds_b128 (ASYNCcnt), s_wait_tensorcnt/asynccnt,
// global_prefetch_b8.

#define BB 4
#define SS 2048
#define DD 1024
#define HH 16
#define HD 64

typedef __bf16 bf16;
typedef __attribute__((ext_vector_type(16))) __bf16 v16bf;
typedef __attribute__((ext_vector_type(8))) float v8f;
typedef int i32x4 __attribute__((ext_vector_type(4)));
typedef int i32x8 __attribute__((ext_vector_type(8)));

// A-matrix fragment (16x32 bf16): lane half 0 holds K=[0..7],[16..23], half 1 holds K=[8..15],[24..31]
__device__ inline v16bf frag_a(const bf16* p, int half) {
  v16bf r;
  ((uint4*)&r)[0] = *(const uint4*)(p + half * 8);
  ((uint4*)&r)[1] = *(const uint4*)(p + 16 + half * 8);
  return r;
}
// B-matrix fragment (32x16 bf16): lane half 0 holds K=[0..7],[16..23], half 1 holds K=[8..15],[24..31]
__device__ inline v16bf frag_b(const bf16* p, int half) {
  v16bf r;
  ((uint4*)&r)[0] = *(const uint4*)(p + half * 8);
  ((uint4*)&r)[1] = *(const uint4*)(p + 16 + half * 8);
  return r;
}

__device__ inline v8f wmma_bf16(v16bf a, v16bf b, v8f c) {
  return __builtin_amdgcn_wmma_f32_16x16x32_bf16(false, a, false, b, (short)0, c,
                                                 false, false);
}

__device__ inline float red_max16(float x) {
  x = fmaxf(x, __shfl_xor(x, 1, 32));
  x = fmaxf(x, __shfl_xor(x, 2, 32));
  x = fmaxf(x, __shfl_xor(x, 4, 32));
  x = fmaxf(x, __shfl_xor(x, 8, 32));
  return x;
}
__device__ inline float red_sum16(float x) {
  x += __shfl_xor(x, 1, 32);
  x += __shfl_xor(x, 2, 32);
  x += __shfl_xor(x, 4, 32);
  x += __shfl_xor(x, 8, 32);
  return x;
}

// Low 32 bits of a shared-aperture generic pointer == LDS byte offset
__device__ inline unsigned lds_off(const void* p) {
  return (unsigned)(uintptr_t)p;
}

// Async global -> LDS copy, 16B per lane, tracked by ASYNCcnt.
__device__ inline void async_copy_b128(unsigned ldsByteOff, const void* gptr) {
  asm volatile("global_load_async_to_lds_b128 %0, %1, off"
               :
               : "v"(ldsByteOff), "v"(gptr)
               : "memory");
}
__device__ inline void wait_asynccnt0() {
  asm volatile("s_wait_asynccnt 0" ::: "memory");
}

// TDM: DMA a 2-D tile (64 rows x 64 bf16, tensor row stride DD elems) from
// global to LDS, padding LDS by 4 DWORDs after every 32 DWORDs (row stride
// 72 elems = 144B). Descriptor groups 0+1 only (2-D tensor). One wave issues.
__device__ inline void tdm_load_k_tile(unsigned ldsByteOff, const void* gtile) {
  const unsigned long long ga = (unsigned long long)gtile;
  unsigned w0[4], w1[8];
  w0[0] = 1u;                                   // count=1 (valid user D#)
  w0[1] = ldsByteOff;                           // lds_addr
  w0[2] = (unsigned)ga;                         // global_addr[31:0]
  w0[3] = (unsigned)((ga >> 32) & 0x1FFFFFFu)   // global_addr[56:32]
          | (2u << 30);                         // type=2 ("image")
  w1[0] = (1u << 16)                            // data_size = 2 bytes
          | (1u << 20)                          // pad_enable
          | (4u << 22)                          // pad_interval: 32 DWORDs
          | (3u << 25);                         // pad_amount: 4 DWORDs
  w1[1] = 64u << 16;                            // tensor_dim0 = 64
  w1[2] = 64u << 16;                            // tensor_dim1 = 64
  w1[3] = 64u << 16;                            // tile_dim0 = 64
  w1[4] = 64u;                                  // tile_dim1 = 64
  w1[5] = (unsigned)DD;                         // tensor_dim0_stride = 1024
  w1[6] = 0u;
  w1[7] = 0u;
  i32x4 g0;
  i32x8 g1;
#pragma unroll
  for (int i = 0; i < 4; ++i) g0[i] = (int)w0[i];
#pragma unroll
  for (int i = 0; i < 8; ++i) g1[i] = (int)w1[i];
  asm volatile("tensor_load_to_lds %0, %1" : : "s"(g0), "s"(g1) : "memory");
}
__device__ inline void wait_tensorcnt0() {
  asm volatile("s_wait_tensorcnt 0x0" ::: "memory");
}

// -------------------------------------------------------------------------
// Y[M x 1024] = A[M x 1024] @ W[1024 x 1024]^T + bias   (torch Linear)
// Block tile 128x128, 8 waves (4 M x 2 N), wave tile 32x64, K step 32.
// A source: f32 (converted in loader) or bf16 (async-copied). W always f32.
// -------------------------------------------------------------------------
template <bool A_F32, bool OUT_F32>
__global__ __launch_bounds__(256) void gemm_bias_kernel(
    const void* __restrict__ Ap, const float* __restrict__ Wp,
    const float* __restrict__ bias, void* __restrict__ Yp) {
  __shared__ bf16 ldsA[128 * 40];  // stride 40 elems (80B, 16B-mult)
  __shared__ bf16 ldsB[128 * 40];

  const int tid = threadIdx.x;
  const int lane = tid & 31;
  const int wave = tid >> 5;
  const int half = lane >> 4;
  const int l16 = lane & 15;
  const int wm = wave & 3;   // wave row 0..3
  const int wn = wave >> 2;  // wave col 0..1
  const int rowBase = blockIdx.x * 128;
  const int colBase = blockIdx.y * 128;

  v8f zero = {};
  v8f acc[2][4];
#pragma unroll
  for (int t = 0; t < 2; ++t)
#pragma unroll
    for (int u = 0; u < 4; ++u) acc[t][u] = zero;

  for (int k0 = 0; k0 < DD; k0 += 32) {
    // prefetch next K-slab into GL2 (global_prefetch_b8)
    if (k0 + 32 < DD) {
      const int r = tid & 127;
      if (A_F32)
        __builtin_prefetch((const float*)Ap + (size_t)(rowBase + r) * DD + k0 + 32, 0, 3);
      else
        __builtin_prefetch((const bf16*)Ap + (size_t)(rowBase + r) * DD + k0 + 32, 0, 3);
      __builtin_prefetch(Wp + (size_t)(colBase + r) * DD + k0 + 32, 0, 3);
    }
    // --- stage tiles into LDS (512 16B-chunks each, 2 per thread) ---
#pragma unroll
    for (int p = 0; p < 2; ++p) {
      const int task = p * 256 + tid;
      const int r = task >> 2;
      const int ch = (task & 3) * 8;
      if (A_F32) {
        const float* s = (const float*)Ap + (size_t)(rowBase + r) * DD + k0 + ch;
        float4 f0 = ((const float4*)s)[0];
        float4 f1 = ((const float4*)s)[1];
        alignas(16) bf16 t8[8] = {(bf16)f0.x, (bf16)f0.y, (bf16)f0.z, (bf16)f0.w,
                                  (bf16)f1.x, (bf16)f1.y, (bf16)f1.z, (bf16)f1.w};
        *(uint4*)&ldsA[r * 40 + ch] = *(const uint4*)t8;
      } else {
        // straight bf16 copy: async global->LDS (ASYNCcnt), bypasses VGPRs
        const bf16* s = (const bf16*)Ap + (size_t)(rowBase + r) * DD + k0 + ch;
        async_copy_b128(lds_off(&ldsA[r * 40 + ch]), s);
      }
      // weight tile: row index = output column (y = x @ W^T)
      const float* sw = Wp + (size_t)(colBase + r) * DD + k0 + ch;
      float4 g0 = ((const float4*)sw)[0];
      float4 g1 = ((const float4*)sw)[1];
      alignas(16) bf16 w8[8] = {(bf16)g0.x, (bf16)g0.y, (bf16)g0.z, (bf16)g0.w,
                                (bf16)g1.x, (bf16)g1.y, (bf16)g1.z, (bf16)g1.w};
      *(uint4*)&ldsB[r * 40 + ch] = *(const uint4*)w8;
    }
    if (!A_F32) wait_asynccnt0();
    __syncthreads();

    v16bf af[2], bfr[4];
#pragma unroll
    for (int t = 0; t < 2; ++t)
      af[t] = frag_a(&ldsA[(wm * 32 + t * 16 + l16) * 40], half);
#pragma unroll
    for (int u = 0; u < 4; ++u)
      bfr[u] = frag_b(&ldsB[(wn * 64 + u * 16 + l16) * 40], half);
#pragma unroll
    for (int t = 0; t < 2; ++t)
#pragma unroll
      for (int u = 0; u < 4; ++u) acc[t][u] = wmma_bf16(af[t], bfr[u], acc[t][u]);
    __syncthreads();
  }

  // epilogue: add bias, store (C/D layout: row m = v + 8*half, col n = l16)
#pragma unroll
  for (int u = 0; u < 4; ++u) {
    const int n = colBase + wn * 64 + u * 16 + l16;
    const float bv = bias[n];
#pragma unroll
    for (int t = 0; t < 2; ++t) {
#pragma unroll
      for (int v = 0; v < 8; ++v) {
        const int m = rowBase + wm * 32 + t * 16 + v + 8 * half;
        const float val = acc[t][u][v] + bv;
        if (OUT_F32)
          ((float*)Yp)[(size_t)m * DD + n] = val;
        else
          ((bf16*)Yp)[(size_t)m * DD + n] = (bf16)val;
      }
    }
  }
}

// -------------------------------------------------------------------------
// Flash attention (causal). One block = (b, h, 128-query strip).
// 8 waves, each owns 16 query rows. Key blocks of 64, online softmax.
// K tile staged by the Tensor Data Mover (TENSORcnt).
// -------------------------------------------------------------------------
__global__ __launch_bounds__(256) void attn_kernel(const bf16* __restrict__ Q,
                                                   const bf16* __restrict__ K,
                                                   const bf16* __restrict__ V,
                                                   bf16* __restrict__ O) {
  __shared__ bf16 ldsK[64 * 72];      // [keypos][hd], stride 72 (TDM pad: +16B/row)
  __shared__ bf16 ldsVt[64 * 72];     // [hd][keypos], stride 72 (pre-transposed)
  __shared__ bf16 ldsP[8][16 * 72];   // per-wave P tile [qrow][keypos]

  const int tid = threadIdx.x;
  const int lane = tid & 31;
  const int wave = tid >> 5;
  const int half = lane >> 4;
  const int l16 = lane & 15;
  const int qBlk = blockIdx.x * 128;
  const int h = blockIdx.y;
  const int bb = blockIdx.z;
  const int qSeq = qBlk + wave * 16;  // this wave's first query row (in-seq)
  const size_t seqBase = (size_t)bb * SS;

  // Q fragments for the whole loop (direct b128 global loads in A-layout)
  const bf16* qp = Q + (seqBase + qSeq + l16) * DD + h * HD;
  v16bf qf[2];
#pragma unroll
  for (int ks = 0; ks < 2; ++ks) qf[ks] = frag_a(qp + ks * 32, half);

  float rm[8], rl[8];
  v8f zero = {};
  v8f oacc[4];
#pragma unroll
  for (int v = 0; v < 8; ++v) { rm[v] = -1e30f; rl[v] = 0.f; }
#pragma unroll
  for (int u = 0; u < 4; ++u) oacc[u] = zero;

  const int nkb = qBlk / 64 + 2;  // causal: key blocks up to strip end
  for (int kb = 0; kb < nkb; ++kb) {
    const int kp0 = kb * 64;

    // --- stage K tile via TDM (one DMA for the whole 64x64 tile) ---
    if (wave == 0) {
      tdm_load_k_tile(lds_off(ldsK), K + (seqBase + kp0) * DD + h * HD);
    }
    // --- stage V tile transposed: ldsVt[hd][kp] ---
#pragma unroll
    for (int i = 0; i < 16; ++i) {
      const int idx = i * 256 + tid;
      const int kp = idx >> 6;
      const int hd = idx & 63;
      ldsVt[hd * 72 + kp] = V[(seqBase + kp0 + kp) * DD + h * HD + hd];
    }
    // prefetch next key block into GL2 while this one is consumed
    if (kb + 1 < nkb) {
      const int kp = tid & 63;
      __builtin_prefetch(K + (seqBase + kp0 + 64 + kp) * DD + h * HD, 0, 3);
      __builtin_prefetch(V + (seqBase + kp0 + 64 + kp) * DD + h * HD, 0, 3);
    }
    if (wave == 0) wait_tensorcnt0();
    __syncthreads();

    // --- scores S = Q K^T (16 x 64 per wave) ---
    v8f s4[4];
#pragma unroll
    for (int u = 0; u < 4; ++u) s4[u] = zero;
#pragma unroll
    for (int ks = 0; ks < 2; ++ks) {
#pragma unroll
      for (int u = 0; u < 4; ++u) {
        v16bf kf = frag_b(&ldsK[(u * 16 + l16) * 72] + ks * 32, half);
        s4[u] = wmma_bf16(qf[ks], kf, s4[u]);
      }
    }

    // --- online softmax (row m = v + 8*half, col = kp0 + u*16 + l16) ---
    bf16* pRow = ldsP[wave];
#pragma unroll
    for (int v = 0; v < 8; ++v) {
      const int qg = qSeq + v + 8 * half;
      float mu[4];
      float rmax = -1e30f;
#pragma unroll
      for (int u = 0; u < 4; ++u) {
        const int kg = kp0 + u * 16 + l16;
        float val = s4[u][v] * 0.125f;  // 1/sqrt(HD)
        val = (kg <= qg) ? val : -1e30f;
        mu[u] = val;
        rmax = fmaxf(rmax, val);
      }
      rmax = red_max16(rmax);
      const float mnew = fmaxf(rm[v], rmax);
      const float alpha = __expf(rm[v] - mnew);
      float rs = 0.f;
#pragma unroll
      for (int u = 0; u < 4; ++u) {
        const int kg = kp0 + u * 16 + l16;
        const float pv = (kg <= qg) ? __expf(mu[u] - mnew) : 0.f;
        pRow[(v + 8 * half) * 72 + u * 16 + l16] = (bf16)pv;
        rs += pv;
      }
      rs = red_sum16(rs);
      rl[v] = rl[v] * alpha + rs;
      rm[v] = mnew;
#pragma unroll
      for (int u = 0; u < 4; ++u) oacc[u][v] *= alpha;
    }

    // --- O += P @ V (A from ldsP, B from ldsVt) ---
#pragma unroll
    for (int ks = 0; ks < 2; ++ks) {
      v16bf pf = frag_a(pRow + l16 * 72 + ks * 32, half);
#pragma unroll
      for (int u = 0; u < 4; ++u) {
        v16bf vf = frag_b(&ldsVt[(u * 16 + l16) * 72] + ks * 32, half);
        oacc[u] = wmma_bf16(pf, vf, oacc[u]);
      }
    }
    __syncthreads();
  }

  // --- normalize and store attn output in [B,S,D] (col = h*HD + hd) ---
#pragma unroll
  for (int v = 0; v < 8; ++v) {
    const float inv = 1.0f / rl[v];
    const size_t row = seqBase + qSeq + v + 8 * half;
#pragma unroll
    for (int u = 0; u < 4; ++u) {
      O[row * DD + h * HD + u * 16 + l16] = (bf16)(oacc[u][v] * inv);
    }
  }
}

// -------------------------------------------------------------------------
extern "C" void kernel_launch(void* const* d_in, const int* in_sizes, int n_in,
                              void* d_out, int out_size, void* d_ws, size_t ws_size,
                              hipStream_t stream) {
  const float* query = (const float*)d_in[0];
  const float* key_in = (const float*)d_in[1];
  const float* value = (const float*)d_in[2];
  // d_in[3] = mask (bool) — causal, computed analytically in-kernel
  const float* Wq = (const float*)d_in[4];
  const float* bq = (const float*)d_in[5];
  const float* Wk = (const float*)d_in[6];
  const float* bk = (const float*)d_in[7];
  const float* Wv = (const float*)d_in[8];
  const float* bv = (const float*)d_in[9];
  const float* Wo = (const float*)d_in[10];
  const float* bo = (const float*)d_in[11];

  const size_t elems = (size_t)BB * SS * DD;  // 8388608
  bf16* Qb = (bf16*)d_ws;
  bf16* Kb = Qb + elems;
  bf16* Vb = Kb + elems;
  bf16* Ab = Vb + elems;  // total 64 MB of workspace

  dim3 gblk(256);
  dim3 ggrid(BB * SS / 128, DD / 128);  // 64 x 8

  // QKV projections (f32 in, bf16 out)
  gemm_bias_kernel<true, false><<<ggrid, gblk, 0, stream>>>(query, Wq, bq, Qb);
  gemm_bias_kernel<true, false><<<ggrid, gblk, 0, stream>>>(key_in, Wk, bk, Kb);
  gemm_bias_kernel<true, false><<<ggrid, gblk, 0, stream>>>(value, Wv, bv, Vb);

  // flash attention
  dim3 agrid(SS / 128, HH, BB);  // 16 x 16 x 4
  attn_kernel<<<agrid, gblk, 0, stream>>>(Qb, Kb, Vb, Ab);

  // output projection (bf16 in, f32 out)
  gemm_bias_kernel<false, true><<<ggrid, gblk, 0, stream>>>(Ab, Wo, bo, (float*)d_out);
}
